// PPLayer_9569187135998
// MI455X (gfx1250) — matrix operand order repaired and checked by reading, not compiled, not run on hardware
//
#include <hip/hip_runtime.h>
#include <hip/hip_bf16.h>

// CDNA5 / gfx1250 wave32 WMMA types
typedef __attribute__((ext_vector_type(16))) _Float16 v16h;
typedef __attribute__((ext_vector_type(8)))  _Float16 v8h;
typedef __attribute__((ext_vector_type(8)))  float    v8f;

// Problem constants
#define BATCH 128
#define NDOM  3
#define NCLS  7
#define NSHOT 10
#define CH    512
#define HW    49          // 7*7
#define DK    128
#define NPOSX (BATCH*HW)              // 6272 query positions
#define NPOSP (NDOM*NCLS*NSHOT*HW)    // 10290 prototype positions
#define NDK   (NDOM*NCLS)             // 21
#define NIJ   (NSHOT*HW)              // 490
#define NIJP  512                     // padded K for second GEMM

// ---- fragment index helpers (per CDNA5 ISA 7.12.2, wave32) ----
// A (16x32 f16): lane L -> M = L%16 ; element j -> K = (L/16)*8 + (j&7) + ((j>>3)<<4)
//   (two contiguous runs of 8: [g*8 .. g*8+7] and [16+g*8 .. 16+g*8+7])
// B (32x16 f16): lane L -> N = L%16 ; this file: A and B both use K = (L/16)*16 + j (contiguous)
// C/D (16x16 f32): lane L, elem r -> M = r + (L/16)*8 ; N = L%16
__device__ __forceinline__ int ka_idx(int g, int j) {
    return g * 8 + (j & 7) + ((j >> 3) << 4);
}

// --------------------------------------------------------------------------
// Zero-fill half buffer (pv_t padding region must be zero for padded-K GEMM)
// --------------------------------------------------------------------------
__global__ void pp_zero_h(_Float16* p, int n) {
    int i = blockIdx.x * 256 + threadIdx.x;
    if (i < n) p[i] = (_Float16)0.f;
}

// --------------------------------------------------------------------------
// Projection of x: q_q (f16, [pos][e]) and q_v (f32, [pos][e])
// pos = b*49 + hw ; x layout (b, c, h, w) -> x[(pos/49)*C*49 + c*49 + pos%49]
// One block = one M-tile of 16 positions; 8 waves = 8 N-tiles of 16 e each.
// --------------------------------------------------------------------------
__global__ __launch_bounds__(256) void pp_proj_x(
    const float* __restrict__ x, const float* __restrict__ wqk,
    const float* __restrict__ wv, _Float16* __restrict__ qq,
    float* __restrict__ qv)
{
    const int tid = threadIdx.x;
    const int wave = tid >> 5, lane = tid & 31;
    const int g = lane >> 4, mr = lane & 15;
    const int mBase = blockIdx.x * 16;

    const int pos = mBase + mr;                 // A-row for this lane
    const int img = pos / HW, ij = pos % HW;
    const float* aBase = x + img * (CH * HW) + ij;
    const int e = wave * 16 + mr;               // B-col for this lane
    const float* wkB = wqk + e * CH;
    const float* wvB = wv  + e * CH;

    v8f accK = {}, accV = {};
    for (int kc = 0; kc < CH / 32; ++kc) {
        v16h a, bk, bv;
#pragma unroll
        for (int j = 0; j < 16; ++j) {
            int c = kc * 32 + g * 16 + j;
            a[j] = (_Float16)aBase[c * HW];
        }
#pragma unroll
        for (int j = 0; j < 16; ++j) {
            int c = kc * 32 + g * 16 + j;
            bk[j] = (_Float16)wkB[c];
            bv[j] = (_Float16)wvB[c];
        }
        accK = __builtin_amdgcn_wmma_f32_16x16x32_f16(false, a, false, bk,
                                                      (short)0, accK, false, false);
        accV = __builtin_amdgcn_wmma_f32_16x16x32_f16(false, a, false, bv,
                                                      (short)0, accV, false, false);
    }
#pragma unroll
    for (int r = 0; r < 8; ++r) {
        int m = r + g * 8;
        int p2 = mBase + m;
        int ee = wave * 16 + mr;
        qq[p2 * DK + ee] = (_Float16)accK[r];
        qv[p2 * DK + ee] = accV[r];
    }
}

// --------------------------------------------------------------------------
// Projection of prototypes: pk (f16, [pp][e]) and pv_t (f16, [dk][e][NIJP])
// pp = dk*490 + nij ; prototypes laid out (d,k,n,c,h,w)
// --------------------------------------------------------------------------
__global__ __launch_bounds__(256) void pp_proj_p(
    const float* __restrict__ prot, const float* __restrict__ wqk,
    const float* __restrict__ wv, _Float16* __restrict__ pk,
    _Float16* __restrict__ pvt)
{
    const int tid = threadIdx.x;
    const int wave = tid >> 5, lane = tid & 31;
    const int g = lane >> 4, mr = lane & 15;
    const int mBase = blockIdx.x * 16;

    const int pos = mBase + mr;
    const bool rowOk = pos < NPOSP;
    const int posc = rowOk ? pos : 0;
    const int img = posc / HW, ij = posc % HW;
    const float* aBase = prot + img * (CH * HW) + ij;
    const int e = wave * 16 + mr;
    const float* wkB = wqk + e * CH;
    const float* wvB = wv  + e * CH;

    v8f accK = {}, accV = {};
    for (int kc = 0; kc < CH / 32; ++kc) {
        v16h a, bk, bv;
#pragma unroll
        for (int j = 0; j < 16; ++j) {
            int c = kc * 32 + g * 16 + j;
            a[j] = rowOk ? (_Float16)aBase[c * HW] : (_Float16)0.f;
        }
#pragma unroll
        for (int j = 0; j < 16; ++j) {
            int c = kc * 32 + g * 16 + j;
            bk[j] = (_Float16)wkB[c];
            bv[j] = (_Float16)wvB[c];
        }
        accK = __builtin_amdgcn_wmma_f32_16x16x32_f16(false, a, false, bk,
                                                      (short)0, accK, false, false);
        accV = __builtin_amdgcn_wmma_f32_16x16x32_f16(false, a, false, bv,
                                                      (short)0, accV, false, false);
    }
#pragma unroll
    for (int r = 0; r < 8; ++r) {
        int m = r + g * 8;
        int p2 = mBase + m;
        if (p2 < NPOSP) {
            int ee = wave * 16 + mr;
            pk[p2 * DK + ee] = (_Float16)accK[r];
            int dk = p2 / NIJ, nij = p2 % NIJ;
            pvt[dk * (DK * NIJP) + ee * NIJP + nij] = (_Float16)accV[r];
        }
    }
}

// --------------------------------------------------------------------------
// Fused: sim (WMMA) -> softmax -> out = attn x pv (WMMA) -> distance.
// One block per (b, dk). 256 threads = 8 waves. Row chunks of 16 (49 rows).
// LDS: 32 KB f32 logits + 16 KB f16 attn.
// --------------------------------------------------------------------------
__global__ __launch_bounds__(256) void pp_fused_attn(
    const _Float16* __restrict__ qq, const float* __restrict__ qv,
    const _Float16* __restrict__ pk, const _Float16* __restrict__ pvt,
    float* __restrict__ out)
{
    __shared__ float simLds[16 * NIJP];                     // logits (f32)
    __shared__ __align__(16) _Float16 attnLds[16 * NIJP];   // attn (f16)
    __shared__ float accDist;

    const int b  = blockIdx.x / NDK;
    const int dk = blockIdx.x % NDK;
    const int tid = threadIdx.x;
    const int wave = tid >> 5, lane = tid & 31;
    const int g = lane >> 4, mr = lane & 15;
    const float scale = 0.08838834764831845f;   // 1/sqrt(128)

    if (tid == 0) accDist = 0.f;

    // Warm the L2/WGP$ with this wave's pv_t panel (global_prefetch_b8):
    // wave handles e-tile [wave*16, wave*16+16); lane's B row is e = wave*16+mr.
    {
        const _Float16* pBase = pvt + (dk * DK + wave * 16 + mr) * NIJP;
#pragma unroll
        for (int c = 0; c < NIJP * 2; c += 256)
            __builtin_prefetch((const char*)pBase + c, 0, 3);
    }
    __syncthreads();

    for (int rc = 0; rc < 4; ++rc) {            // 4 chunks of 16 query rows
        const int rowBase = rc * 16;

        // zero the K-pad columns (490..511) of the f16 attn buffer
        for (int idx = tid; idx < 16 * (NIJP - NIJ); idx += 256) {
            attnLds[(idx / (NIJP - NIJ)) * NIJP + NIJ + (idx % (NIJP - NIJ))] =
                (_Float16)0.f;
        }

        // ---- Phase A: sim chunk = q_q(16x128) x pk^T(128x490) ----
        // Hoist the 4 A fragments (depend only on rc/kc, not the N tile).
        const int qrow = rowBase + mr;          // A-row: query spatial pos
        const bool rowOk = qrow < HW;
        const _Float16* aBase = qq + (b * HW + (rowOk ? qrow : 0)) * DK;
        v16h aF[DK / 32];
#pragma unroll
        for (int kc = 0; kc < DK / 32; ++kc) {
#pragma unroll
            for (int j = 0; j < 16; ++j)
                aF[kc][j] = rowOk ? aBase[kc * 32 + g * 16 + j] : (_Float16)0.f;
        }

        for (int nt = wave; nt < 31; nt += 8) {
            v8f acc = {};
            const int nij = nt * 16 + mr;       // B-col: prototype position
            const bool colOk = nij < NIJ;
            const _Float16* bBase = pk + (dk * NIJ + nij) * DK;
#pragma unroll
            for (int kc = 0; kc < DK / 32; ++kc) {
                v16h bb = {};
                if (colOk)   // contiguous 32B per lane: pk row-major [pp][e]
                    bb = *(const v16h*)(bBase + kc * 32 + g * 16);
                acc = __builtin_amdgcn_wmma_f32_16x16x32_f16(false, aF[kc], false, bb,
                                                             (short)0, acc, false, false);
            }
#pragma unroll
            for (int r = 0; r < 8; ++r) {
                int row = r + g * 8;
                int col = nt * 16 + mr;
                if (col < NIJ) simLds[row * NIJP + col] = acc[r] * scale;
            }
        }
        __syncthreads();

        // ---- Phase B: softmax over 490 cols; 16 threads per row.
        //      Writes normalized attn directly as f16 into attnLds. ----
        {
            const int row = tid >> 4;
            const int lg  = tid & 15;
            float mx = -1e30f;
            for (int c = lg; c < NIJ; c += 16) mx = fmaxf(mx, simLds[row * NIJP + c]);
            for (int off = 8; off; off >>= 1) mx = fmaxf(mx, __shfl_xor(mx, off, 16));
            float s = 0.f;
            for (int c = lg; c < NIJ; c += 16) {
                float e = __expf(simLds[row * NIJP + c] - mx);
                simLds[row * NIJP + c] = e;
                s += e;
            }
            for (int off = 8; off; off >>= 1) s += __shfl_xor(s, off, 16);
            const float inv = 1.f / s;
            for (int c = lg; c < NIJ; c += 16)
                attnLds[row * NIJP + c] = (_Float16)(simLds[row * NIJP + c] * inv);
        }
        __syncthreads();

        // ---- Phase C: out chunk = attn(16x512pad) x pv(512x128); wave -> e-tile ----
        // A fragments now come as two aligned 16-byte LDS vector loads.
        v8f acc = {};
        {
            const _Float16* bBase = pvt + (dk * DK + wave * 16 + mr) * NIJP;
            const _Float16* aRow  = attnLds + mr * NIJP;
#pragma unroll
            for (int kc = 0; kc < NIJP / 32; ++kc) {
                const v8h lo = *(const v8h*)(aRow + kc * 32 + g * 16);
                const v8h hi = *(const v8h*)(aRow + kc * 32 + g * 16 + 8);
                v16h a;
#pragma unroll
                for (int j = 0; j < 8; ++j) { a[j] = lo[j]; a[j + 8] = hi[j]; }
                v16h bb = *(const v16h*)(bBase + kc * 32 + g * 16);
                acc = __builtin_amdgcn_wmma_f32_16x16x32_f16(false, a, false, bb,
                                                             (short)0, acc, false, false);
            }
        }

        // ---- Phase D: squared distance vs q_v ----
        float dsq = 0.f;
#pragma unroll
        for (int r = 0; r < 8; ++r) {
            int row = rowBase + r + g * 8;
            if (row < HW) {
                int e = wave * 16 + mr;
                float diff = qv[(b * HW + row) * DK + e] - acc[r];
                dsq += diff * diff;
            }
        }
        for (int off = 16; off; off >>= 1) dsq += __shfl_xor(dsq, off, 32);
        if (lane == 0) atomicAdd(&accDist, dsq);
        __syncthreads();
    }

    if (tid == 0) out[b * NDK + dk] = -accDist * (1.0f / (float)HW);
}

// --------------------------------------------------------------------------
extern "C" void kernel_launch(void* const* d_in, const int* in_sizes, int n_in,
                              void* d_out, int out_size, void* d_ws, size_t ws_size,
                              hipStream_t stream) {
    (void)in_sizes; (void)n_in; (void)out_size; (void)ws_size;
    const float* x    = (const float*)d_in[0];
    const float* prot = (const float*)d_in[1];
    const float* wqk  = (const float*)d_in[2];
    const float* wv   = (const float*)d_in[3];
    float* out = (float*)d_out;

    // workspace carve (all offsets 256B aligned)
    char* ws = (char*)d_ws;
    _Float16* qq  = (_Float16*)(ws + 0);                 // 6272*128 f16 = 1,605,632 B
    float*    qv  = (float*)   (ws + 1605632);           // 6272*128 f32 = 3,211,264 B
    _Float16* pk  = (_Float16*)(ws + 4816896);           // 10290*128 f16 = 2,634,240 B
    _Float16* pvt = (_Float16*)(ws + 7451136);           // 21*128*512 f16 = 2,752,512 B

    const int pvtN = NDK * DK * NIJP;                    // 1,376,256 halves
    pp_zero_h<<<(pvtN + 255) / 256, 256, 0, stream>>>(pvt, pvtN);
    pp_proj_x<<<NPOSX / 16, 256, 0, stream>>>(x, wqk, wv, qq, qv);          // 392 blocks
    pp_proj_p<<<(NPOSP + 15) / 16, 256, 0, stream>>>(prot, wqk, wv, pk, pvt); // 644 blocks
    pp_fused_attn<<<BATCH * NDK, 256, 0, stream>>>(qq, qv, pk, pvt, out);   // 2688 blocks
}
